// DirectMultiHorizonHyperbolicForecaster_28252294873665
// MI455X (gfx1250) — matrix operand order repaired and checked by reading, not compiled, not run on hardware
//
#include <hip/hip_runtime.h>
#include <hip/hip_bf16.h>
#include <math.h>

typedef __attribute__((ext_vector_type(16))) _Float16 v16h;
typedef __attribute__((ext_vector_type(8)))  float    v8f;

#define EPSf  1e-5f
#define MAXNf (1.0f - 1e-5f)
#define Bn 32
#define Ln 336
#define Fn 256
#define Dn 128
#define Hn 256
#define SEGn 24
#define Nn 14
#define Tn 4
#define BF 8192            // B*F
#define ROWS4 32768        // 4*BF  and also BF*T

// B-layout forward map: lane, j -> k within 32-K tile
//   k = ((lane>>4)<<4) + j
// A-layout forward map: lane, j -> (m = lane&15, k = ((lane>>4)<<4) + j)
// A-layout inverse map: (m, koff) -> element index within 512-entry tile
__device__ __forceinline__ int amap(int m, int koff) {
    int hi = (koff >> 4) & 1;
    int j  = koff & 15;
    return (m + (hi << 4)) * 16 + j;
}

// ---------------------------------------------------------------- kernel 0
// One-time per launch: convert weights to f16 pre-swizzled into WMMA B-layout.
// bEnc: [4 comp][8 ntile][512]   (K=24 padded to 32)
// bW1 : [4 ktile][16 ntile][512]
// bW2 : [8 ktile][2 ntile][512]  (N=24 padded to 32)
__global__ void __launch_bounds__(256) k_prepw(
    const float* __restrict__ encW, const float* __restrict__ W1,
    const float* __restrict__ W2,
    _Float16* __restrict__ bEnc, _Float16* __restrict__ bW1,
    _Float16* __restrict__ bW2)
{
    int e = blockIdx.x * 256 + threadIdx.x;   // 224 blocks -> 57344 entries
    if (e < 16384) {                          // enc_W
        int c = e >> 12, rem = e & 4095;
        int nt = rem >> 9, pos = rem & 511;
        int lane = pos >> 4, j = pos & 15;
        int k = ((lane >> 4) << 4) + j;
        int n = nt * 16 + (lane & 15);
        float v = (k < SEGn) ? encW[(c * SEGn + k) * Dn + n] : 0.f;
        bEnc[e] = (_Float16)v;
    } else if (e < 16384 + 32768) {           // rec_W1
        int e1 = e - 16384;
        int kt = e1 >> 13, rem = e1 & 8191;
        int nt = rem >> 9, pos = rem & 511;
        int lane = pos >> 4, j = pos & 15;
        int k = kt * 32 + ((lane >> 4) << 4) + j;
        int n = nt * 16 + (lane & 15);
        bW1[e1] = (_Float16)W1[k * Hn + n];
    } else {                                  // rec_W2
        int e2 = e - 16384 - 32768;
        int kt = e2 >> 10, rem = e2 & 1023;
        int nt = rem >> 9, pos = rem & 511;
        int lane = pos >> 4, j = pos & 15;
        int k = kt * 32 + ((lane >> 4) << 4) + j;
        int col = nt * 16 + (lane & 15);
        float v = (col < SEGn) ? W2[k * SEGn + col] : 0.f;
        bW2[e2] = (_Float16)v;
    }
}

// ---------------------------------------------------------------- kernel 1
// RevIN stats: one wave per (b,f) row; also zero the loss accumulator.
__global__ void __launch_bounds__(256) k_stats(
    const float* __restrict__ t0, const float* __restrict__ t1,
    const float* __restrict__ t2, const float* __restrict__ t3,
    float* __restrict__ meanw, float* __restrict__ stdw,
    float* __restrict__ out)
{
    if (blockIdx.x == 0 && threadIdx.x == 0) out[(size_t)Bn*96*Fn] = 0.f;
    int wave = threadIdx.x >> 5, lane = threadIdx.x & 31;
    int r = blockIdx.x * 8 + wave;           // 1024 blocks * 8 waves = 8192
    int b = r >> 8, f = r & 255;
    float s = 0.f, sq = 0.f;
    for (int l = lane; l < Ln; l += 32) {
        size_t idx = ((size_t)(b * Ln + l)) * Fn + f;
        float x = t0[idx] + t1[idx] + t2[idx] + t3[idx];
        s += x; sq += x * x;
    }
    #pragma unroll
    for (int off = 16; off > 0; off >>= 1) {
        s  += __shfl_xor(s,  off, 32);
        sq += __shfl_xor(sq, off, 32);
    }
    if (lane == 0) {
        float m = s / (float)Ln;
        meanw[r] = m;
        stdw[r]  = sqrtf(sq / (float)Ln - m * m + 1e-5f);
    }
}

// ---------------------------------------------------------------- helpers
__device__ __forceinline__ float blockReduce128(float v, float* red, int tid)
{
    red[tid] = v; __syncthreads();
    #pragma unroll
    for (int s = 64; s > 0; s >>= 1) {
        if (tid < s) red[tid] += red[tid + s];
        __syncthreads();
    }
    float r = red[0];
    __syncthreads();
    return r;
}

// ---------------------------------------------------------------- kernel 2
// One block per (component c, row r in BF). Full hyperbolic branch.
__global__ void __launch_bounds__(128) k_branch(
    const float* __restrict__ t0, const float* __restrict__ t1,
    const float* __restrict__ t2, const float* __restrict__ t3,
    const float* __restrict__ revw, const float* __restrict__ revb,
    const _Float16* __restrict__ bEnc, const float* __restrict__ encB,
    const float* __restrict__ velW, const float* __restrict__ velB,
    const float* __restrict__ steps,
    const float* __restrict__ meanw, const float* __restrict__ stdw,
    float* __restrict__ zfut, float* __restrict__ dists)
{
    __shared__ _Float16 sAh[512];          // 16x32 A tile, WMMA A-layout (contiguous k = 16*(lane>>4) + j)
    __shared__ float zb[16 * Dn];          // latents z (rows 0..13 valid)
    __shared__ float ub[13 * Dn];          // tangent vectors u
    __shared__ float red[128];
    __shared__ float sx2[16], sxy[16], ssc[16];
    __shared__ float su0[128];

    int tid = threadIdx.x;
    int c = blockIdx.x >> 13;              // component 0..3
    int r = blockIdx.x & (BF - 1);
    int b = r >> 8, f = r & 255;
    const float* comp = (c == 0) ? t0 : (c == 1) ? t1 : (c == 2) ? t2 : t3;

    float mean = meanw[r], sd = stdw[r];
    float rw = revw[f], rb = revb[f];

    __builtin_prefetch(&velW[((size_t)c * Dn) * Dn + tid], 0, 1);

    // ---- phase 1a: zero the padded A tile
    for (int e = tid; e < 512; e += 128) sAh[e] = (_Float16)0.f;
    __syncthreads();
    // ---- phase 1b: unconditional gather of the 336 valid elements, swizzled store
    for (int e = tid; e < Nn * SEGn; e += 128) {
        int m = e / SEGn, k = e - m * SEGn;
        float raw = comp[((size_t)(b * Ln + e)) * Fn + f];
        float val = (raw - 0.25f * mean) / sd * rw + 0.25f * rb;
        sAh[amap(m, k)] = (_Float16)val;
    }
    __syncthreads();

    // ---- phase 2: encoder GEMM [16x32]@[32x128] via WMMA (bias in accumulator)
    {
        int wave = tid >> 5, lane = tid & 31;
        v16h a = *(const v16h*)&sAh[lane * 16];
        #pragma unroll
        for (int tt = 0; tt < 2; ++tt) {
            int nt  = wave * 2 + tt;
            int col = nt * 16 + (lane & 15);
            v16h bm = *(const v16h*)&bEnc[((size_t)c * 8 + nt) * 512 + lane * 16];
            float bias = encB[c * Dn + col];
            v8f acc;
            #pragma unroll
            for (int q = 0; q < 8; ++q) acc[q] = bias;
            acc = __builtin_amdgcn_wmma_f32_16x16x32_f16(
                      false, a, false, bm, (short)0, acc, false, false);
            #pragma unroll
            for (int q = 0; q < 8; ++q) {
                int mrow = q + ((lane >> 4) << 3);
                zb[mrow * Dn + col] = acc[q];
            }
        }
    }
    __syncthreads();

    // ---- phase 3: expmap0 + projx per segment row
    if (tid < Nn) {
        float ss = 0.f;
        for (int d = 0; d < Dn; ++d) { float z = zb[tid * Dn + d]; ss += z * z; }
        float vn = sqrtf(fmaxf(ss, EPSf * EPSf));
        float th = tanhf(vn);
        float sc = th / vn;
        if (th > MAXNf) sc = MAXNf / vn;
        ssc[tid] = sc;
    }
    __syncthreads();
    for (int e = tid; e < Nn * Dn; e += 128) zb[e] *= ssc[e >> 7];
    __syncthreads();

    // ---- phase 4: per-row norms and neighbor dot products
    if (tid < Nn) {
        float ss = 0.f;
        for (int d = 0; d < Dn; ++d) { float z = zb[tid * Dn + d]; ss += z * z; }
        sx2[tid] = ss;
    } else if (tid >= 16 && tid < 16 + (Nn - 1)) {
        int n = tid - 16;
        float ss = 0.f;
        for (int d = 0; d < Dn; ++d) ss += zb[n * Dn + d] * zb[(n + 1) * Dn + d];
        sxy[n] = ss;
    }
    __syncthreads();

    // ---- phase 5: u[n] = mobius_add(-z[n], z[n+1])
    for (int e = tid; e < (Nn - 1) * Dn; e += 128) {
        int n = e >> 7, d = e & 127;
        float x = zb[n * Dn + d], y = zb[(n + 1) * Dn + d];
        float X2 = sx2[n], Y2 = sx2[n + 1], XY = sxy[n];
        float num = (1.f - 2.f * XY + Y2) * (-x) + (1.f - X2) * y;
        float den = fmaxf(1.f - 2.f * XY + X2 * Y2, EPSf);
        ub[e] = num / den;
    }
    __syncthreads();

    // ---- phase 6: logmap coefficient * decay weight per n
    if (tid < Nn - 1) {
        float ss = 0.f;
        for (int d = 0; d < Dn; ++d) { float u = ub[tid * Dn + d]; ss += u * u; }
        float un   = sqrtf(fmaxf(ss, EPSf * EPSf));
        float coef = fmaxf(1.f - sx2[tid], EPSf) * atanhf(fminf(un, MAXNf)) / un;
        float Z    = (1.f - powf(0.9f, 13.f)) / 0.1f;
        float w    = powf(0.9f, (float)(12 - tid)) / Z / 13.f;
        ssc[tid] = coef * w;
    }
    __syncthreads();

    // ---- phase 7: v_init (thread owns one d) -> logmap0
    float vi = 0.f;
    for (int n = 0; n < Nn - 1; ++n) vi += ssc[n] * ub[n * Dn + tid];
    float nvi2 = blockReduce128(vi * vi, red, tid);
    float nvi  = sqrtf(fmaxf(nvi2, EPSf * EPSf));
    su0[tid] = atanhf(fminf(nvi, MAXNf)) / nvi * vi;
    __syncthreads();

    // ---- phase 8: PoincareLinear 128x128 GEMV + expmap0 + projx
    float acc = velB[c * Dn + tid];
    for (int din = 0; din < Dn; ++din)
        acc += su0[din] * velW[((size_t)c * Dn + din) * Dn + tid];
    float nv2 = blockReduce128(acc * acc, red, tid);
    float vnn = sqrtf(fmaxf(nv2, EPSf * EPSf));
    float th  = tanhf(vnn);
    float sc  = th / vnn;
    if (th > MAXNf) sc = MAXNf / vnn;
    float v0    = sc * acc;
    float nv0sq = blockReduce128(v0 * v0, red, tid);
    float dzx   = blockReduce128(zb[13 * Dn + tid] * v0, red, tid);

    // ---- phase 9: four horizons of expmap from z[13]
    float x2l = sx2[Nn - 1];
    float sig = 1.f / (1.f + expf(-steps[c]));
    float xlast = zb[13 * Dn + tid];
    for (int t = 1; t <= Tn; ++t) {
        float st  = sig * (float)t;
        float vnt = sqrtf(fmaxf(st * st * nv0sq, EPSf * EPSf));
        float g   = tanhf(vnt / fmaxf(1.f - x2l, EPSf));
        float wd  = g * st * v0 / vnt;
        float xw  = g * st * dzx / vnt;
        float w2  = g * g;
        float num = (1.f + 2.f * xw + w2) * xlast + (1.f - x2l) * wd;
        float den = fmaxf(1.f + 2.f * xw + x2l * w2, EPSf);
        float res = num / den;
        float nr2 = blockReduce128(res * res, red, tid);
        float nr  = sqrtf(fmaxf(nr2, EPSf * EPSf));
        if (nr > MAXNf) res *= MAXNf / nr;
        zfut[(((size_t)c * BF + r) * Tn + (t - 1)) * Dn + tid] = res;
        if (tid == 0) {
            float nn = fminf(nr, MAXNf);
            dists[((size_t)c * BF + r) * Tn + (t - 1)] = 2.f * atanhf(nn);
        }
    }
}

// ---------------------------------------------------------------- kernel 3
// 16 (r,t) rows per block: Mobius fusion + logmap0, then WMMA MLP head.
__global__ void __launch_bounds__(128) k_head(
    const float* __restrict__ zfut, const float* __restrict__ mobw,
    const _Float16* __restrict__ bW1, const float* __restrict__ b1,
    const _Float16* __restrict__ bW2, const float* __restrict__ b2,
    const float* __restrict__ revw, const float* __restrict__ revb,
    const float* __restrict__ meanw, const float* __restrict__ stdw,
    float* __restrict__ out)
{
    __shared__ _Float16 uA[4 * 512];   // u rows in WMMA A-layout (4 k-tiles)
    __shared__ _Float16 hA[8 * 512];   // relu(h) rows in WMMA A-layout (8 k-tiles)
    __shared__ float red2[16][8];
    __shared__ float w4[4];

    int tid = threadIdx.x;
    if (tid == 0) {
        float m = fmaxf(fmaxf(mobw[0], mobw[1]), fmaxf(mobw[2], mobw[3]));
        float e0 = expf(mobw[0] - m), e1 = expf(mobw[1] - m);
        float e2 = expf(mobw[2] - m), e3 = expf(mobw[3] - m);
        float s = e0 + e1 + e2 + e3;
        w4[0] = e0 / s; w4[1] = e1 / s; w4[2] = e2 / s; w4[3] = e3 / s;
    }
    __syncthreads();

    int row = tid >> 3, l8 = tid & 7;  // 8 threads per fused row, 16 d each
    size_t g = (size_t)blockIdx.x * 16 + row;

    auto rowsum = [&](float p) -> float {
        __syncthreads();
        red2[row][l8] = p;
        __syncthreads();
        float s = 0.f;
        #pragma unroll
        for (int i = 0; i < 8; ++i) s += red2[row][i];
        return s;
    };
    auto load16 = [&](const float* base, float* dst) {
        const float4* p4 = (const float4*)base;
        #pragma unroll
        for (int q = 0; q < 4; ++q) {
            float4 v = p4[q];
            dst[q*4+0] = v.x; dst[q*4+1] = v.y; dst[q*4+2] = v.z; dst[q*4+3] = v.w;
        }
    };

    // ---- Mobius fusion: comb = smul(w0,z0) (+) smul(w1,z1) (+) ...
    float comb[16], y[16], p;
    {
        load16(zfut + g * Dn + l8 * 16, y);
        p = 0.f;
        #pragma unroll
        for (int i = 0; i < 16; ++i) p += y[i] * y[i];
        float n2 = rowsum(p);
        float xn = sqrtf(fmaxf(n2, EPSf * EPSf));
        float tn = tanhf(w4[0] * atanhf(fminf(xn, MAXNf)));
        float sc = fminf(tn, MAXNf) / xn;           // mob_smul incl. projx
        #pragma unroll
        for (int i = 0; i < 16; ++i) comb[i] = sc * y[i];
    }
    for (int cc = 1; cc < 4; ++cc) {
        load16(zfut + ((size_t)cc * ROWS4 + g) * Dn + l8 * 16, y);
        p = 0.f;
        #pragma unroll
        for (int i = 0; i < 16; ++i) p += y[i] * y[i];
        float yn2 = rowsum(p);
        float yn  = sqrtf(fmaxf(yn2, EPSf * EPSf));
        float tnn = tanhf(w4[cc] * atanhf(fminf(yn, MAXNf)));
        float scc = fminf(tnn, MAXNf) / yn;
        #pragma unroll
        for (int i = 0; i < 16; ++i) y[i] *= scc;
        float Y2 = scc * scc * yn2;
        p = 0.f;
        #pragma unroll
        for (int i = 0; i < 16; ++i) p += comb[i] * comb[i];
        float X2 = rowsum(p);
        p = 0.f;
        #pragma unroll
        for (int i = 0; i < 16; ++i) p += comb[i] * y[i];
        float XY = rowsum(p);
        float den = fmaxf(1.f + 2.f * XY + X2 * Y2, EPSf);
        float ca = (1.f + 2.f * XY + Y2), cb = (1.f - X2);
        #pragma unroll
        for (int i = 0; i < 16; ++i) comb[i] = (ca * comb[i] + cb * y[i]) / den;
    }
    // projx + logmap0 -> u, stored straight into WMMA A-layout (f16)
    p = 0.f;
    #pragma unroll
    for (int i = 0; i < 16; ++i) p += comb[i] * comb[i];
    float cn2 = rowsum(p);
    float cn  = sqrtf(fmaxf(cn2, EPSf * EPSf));
    float psc = (cn > MAXNf) ? MAXNf / cn : 1.f;
    float xn  = fminf(cn, MAXNf);
    float tot = psc * atanhf(xn) / xn;
    #pragma unroll
    for (int i = 0; i < 16; ++i) {
        int d = l8 * 16 + i;
        uA[(d >> 5) * 512 + amap(row, d & 31)] = (_Float16)(tot * comb[i]);
    }
    __syncthreads();

    // ---- GEMM1: [16x128]@[128x256] + b1, relu -> hA (WMMA, 4 waves x 4 ntiles)
    int wave = tid >> 5, lane = tid & 31;
    int cl = lane & 15, hi = lane >> 4;
    {
        v8f accs[4];
        int cols[4];
        #pragma unroll
        for (int i = 0; i < 4; ++i) {
            int nt = wave + 4 * i;
            cols[i] = nt * 16 + cl;
            float bias = b1[cols[i]];
            #pragma unroll
            for (int q = 0; q < 8; ++q) accs[i][q] = bias;
        }
        for (int kt = 0; kt < 4; ++kt) {
            v16h av = *(const v16h*)&uA[kt * 512 + lane * 16];
            #pragma unroll
            for (int i = 0; i < 4; ++i) {
                int nt = wave + 4 * i;
                v16h bv = *(const v16h*)&bW1[((size_t)kt * 16 + nt) * 512 + lane * 16];
                accs[i] = __builtin_amdgcn_wmma_f32_16x16x32_f16(
                              false, av, false, bv, (short)0, accs[i], false, false);
            }
        }
        #pragma unroll
        for (int i = 0; i < 4; ++i)
            #pragma unroll
            for (int q = 0; q < 8; ++q) {
                int m = q + (hi << 3);
                int n = cols[i];
                hA[(n >> 5) * 512 + amap(m, n & 31)] =
                    (_Float16)fmaxf(accs[i][q], 0.f);
            }
    }
    __syncthreads();

    // ---- GEMM2: [16x256]@[256x24] + b2 -> denorm + store (waves 0,1)
    if (wave < 2) {
        int col = wave * 16 + cl;
        bool valid = (col < SEGn);
        v8f acc;
        float bias = valid ? b2[col] : 0.f;
        #pragma unroll
        for (int q = 0; q < 8; ++q) acc[q] = bias;
        for (int kt = 0; kt < 8; ++kt) {
            v16h av = *(const v16h*)&hA[kt * 512 + lane * 16];
            v16h bv = *(const v16h*)&bW2[((size_t)kt * 2 + wave) * 512 + lane * 16];
            acc = __builtin_amdgcn_wmma_f32_16x16x32_f16(
                      false, av, false, bv, (short)0, acc, false, false);
        }
        if (valid) {
            #pragma unroll
            for (int q = 0; q < 8; ++q) {
                int m = q + (hi << 3);
                size_t gg = (size_t)blockIdx.x * 16 + m;
                int rr = (int)(gg >> 2), tt = (int)(gg & 3);
                int bb = rr >> 8, ff = rr & 255;
                float val = (acc[q] - revb[ff]) / revw[ff] * stdw[rr] + meanw[rr];
                out[((size_t)bb * 96 + (tt * SEGn + col)) * Fn + ff] = val;
            }
        }
    }
}

// ---------------------------------------------------------------- kernel 4
__global__ void __launch_bounds__(256) k_loss(
    const float* __restrict__ dists, float* __restrict__ out)
{
    __shared__ float red[256];
    int i = blockIdx.x * 256 + threadIdx.x;      // 128 blocks cover 32768
    float d0 = dists[i];
    float d1 = dists[ROWS4 + i];
    float d2 = dists[2 * ROWS4 + i];
    float d3 = dists[3 * ROWS4 + i];
    float v = fmaxf(d0 - d1 + 0.1f, 0.f) + fmaxf(d1 - d2 + 0.1f, 0.f)
            + fmaxf(d2 - d3 + 0.1f, 0.f);
    red[threadIdx.x] = v; __syncthreads();
    #pragma unroll
    for (int s = 128; s > 0; s >>= 1) {
        if (threadIdx.x < s) red[threadIdx.x] += red[threadIdx.x + s];
        __syncthreads();
    }
    if (threadIdx.x == 0)
        atomicAdd(out + (size_t)Bn * 96 * Fn, red[0] * (1.f / (float)ROWS4));
}

// ---------------------------------------------------------------- launch
extern "C" void kernel_launch(void* const* d_in, const int* in_sizes, int n_in,
                              void* d_out, int out_size, void* d_ws, size_t ws_size,
                              hipStream_t stream) {
    const float* trend = (const float*)d_in[0];
    const float* seasc = (const float*)d_in[1];
    const float* seasf = (const float*)d_in[2];
    const float* resid = (const float*)d_in[3];
    const float* revw  = (const float*)d_in[4];
    const float* revb  = (const float*)d_in[5];
    const float* encW  = (const float*)d_in[6];
    const float* encB  = (const float*)d_in[7];
    const float* velW  = (const float*)d_in[8];
    const float* velB  = (const float*)d_in[9];
    const float* steps = (const float*)d_in[10];
    const float* mobw  = (const float*)d_in[11];
    const float* W1    = (const float*)d_in[12];
    const float* b1    = (const float*)d_in[13];
    const float* W2    = (const float*)d_in[14];
    const float* b2    = (const float*)d_in[15];
    float* out = (float*)d_out;

    float* ws      = (float*)d_ws;
    float* w_mean  = ws;                                   // 8192
    float* w_std   = ws + BF;                              // 8192
    float* w_zfut  = ws + 2 * BF;                          // 4*8192*4*128
    float* w_dist  = w_zfut + (size_t)4 * BF * Tn * Dn;    // 131072
    _Float16* w_bEnc = (_Float16*)(w_dist + ROWS4 * 4);    // 16384 halfs
    _Float16* w_bW1  = w_bEnc + 16384;                     // 32768 halfs
    _Float16* w_bW2  = w_bW1 + 32768;                      // 8192 halfs

    k_prepw<<<224, 256, 0, stream>>>(encW, W1, W2, w_bEnc, w_bW1, w_bW2);
    k_stats<<<1024, 256, 0, stream>>>(trend, seasc, seasf, resid,
                                      w_mean, w_std, out);
    k_branch<<<4 * BF, 128, 0, stream>>>(trend, seasc, seasf, resid,
                                         revw, revb, w_bEnc, encB, velW, velB,
                                         steps, w_mean, w_std, w_zfut, w_dist);
    k_head<<<ROWS4 / 16, 128, 0, stream>>>(w_zfut, mobw, w_bW1, b1, w_bW2, b2,
                                           revw, revb, w_mean, w_std, out);
    k_loss<<<ROWS4 / 256, 256, 0, stream>>>(w_dist, out);
}
